// lstm_crf_22995254903357
// MI455X (gfx1250) — matrix operand order repaired and checked by reading, not compiled, not run on hardware
//
#include <hip/hip_runtime.h>
#include <stdint.h>

// ---------------------------------------------------------------------------
// BiLSTM(2-layer, H=500/dir) + CRF NLL for MI455X (gfx1250, wave32, WMMA).
// bf16 WMMA (f32 accum) everywhere; input-side gate GEMMs precomputed in bulk;
// per-timestep recurrent GEMM fuses precomputed gates via WMMA C-preload.
// GEMM: one wave computes a 16x64 output tile (4 accumulators, shared A frag)
// for ~1.6x higher arithmetic intensity than one-tile-per-wave.
// ---------------------------------------------------------------------------

#define BZ   64      // batch
#define SS   256     // sequence length
#define EE   300     // embed dim
#define EP   320     // embed dim padded to 32
#define HH   500     // hidden per direction
#define HP   512     // padded hidden
#define GG   2000    // 4*H gates (real)
#define GGP  2048    // gate dim padded to 64-wide N tiling
#define D2   1000    // 2*H
#define D2P  1024    // padded 2*H
#define NT   64      // tags
#define MM   (BZ*SS) // 16384 rows, time-major: row = t*BZ + b

typedef __bf16 v16bf __attribute__((ext_vector_type(16)));
typedef float  v8f   __attribute__((ext_vector_type(8)));
union Frag { v16bf v; uint4 u[2]; };

// ---------------------------------------------------------------------------
// utility kernels
// ---------------------------------------------------------------------------
__global__ void zero_u32_kernel(uint32_t* p, size_t n) {
    size_t stride = (size_t)gridDim.x * blockDim.x;
    for (size_t i = (size_t)blockIdx.x * blockDim.x + threadIdx.x; i < n; i += stride)
        p[i] = 0u;
}

// fp32 [Nreal,K] row-major -> bf16 [Npad,Kp] row-major, zero outside real region
__global__ void cvt_pad_bf16_kernel(const float* __restrict__ src, __bf16* __restrict__ dst,
                                    int Nreal, int Npad, int K, int Kp) {
    size_t tot = (size_t)Npad * Kp;
    size_t stride = (size_t)gridDim.x * blockDim.x;
    for (size_t i = (size_t)blockIdx.x * blockDim.x + threadIdx.x; i < tot; i += stride) {
        int n = (int)(i / Kp), kk = (int)(i % Kp);
        float v = (n < Nreal && kk < K) ? src[(size_t)n * K + kk] : 0.0f;
        dst[i] = (__bf16)v;
    }
}

// gather embeddings: dst [MM, EP] bf16, row = t*BZ + b, token = x[b*SS + t]
__global__ void embed_gather_kernel(const int* __restrict__ x,
                                    const float* __restrict__ embed,
                                    __bf16* __restrict__ dst) {
    size_t tot = (size_t)MM * EP;
    size_t stride = (size_t)gridDim.x * blockDim.x;
    for (size_t i = (size_t)blockIdx.x * blockDim.x + threadIdx.x; i < tot; i += stride) {
        int row = (int)(i / EP), k = (int)(i % EP);
        int t = row >> 6, b = row & 63;          // BZ = 64
        int tok = x[b * SS + t];
        float v = (k < EE) ? embed[(size_t)tok * EE + k] : 0.0f;
        dst[i] = (__bf16)v;
    }
}

// ---------------------------------------------------------------------------
// bf16 WMMA GEMM:  Cout[M,N] = A[M,Kp] @ Wt[N,Kp]^T (+Cin) (+bias) (*mask)
//   - A, Wt row-major bf16; Kp multiple of 32; N multiple of 64
//   - one wave computes a 16x64 output tile: 4 WMMA accumulators share the
//     A fragment each K-step (10 b128 loads per 4 wmma ops)
//   - Cin (optional): fp32 [M,N] preloaded into the accumulators
//   - bias (optional): fp32 [N]
//   - tok (optional): int [BZ,SS]; row m -> b=m&63, t=m>>6; mask = tok>0
// ---------------------------------------------------------------------------
__global__ void __launch_bounds__(128)
gemm_bf16_wmma_kernel(const __bf16* __restrict__ A, const __bf16* __restrict__ Wt,
                      const float* __restrict__ Cin, float* __restrict__ Cout,
                      const float* __restrict__ bias, const int* __restrict__ tok,
                      int M, int N, int Kp) {
    const int lane = threadIdx.x & 31;
    const int wave = threadIdx.x >> 5;
    const int groupsN = N >> 6;                       // 64-wide super-tiles
    const int total   = (M >> 4) * groupsN;
    int g = blockIdx.x * 4 + wave;
    if (g >= total) return;                           // uniform per wave
    const int m0 = (g / groupsN) << 4;
    const int n0 = (g % groupsN) << 6;
    const int l16 = lane & 15;
    const int kh  = lane >> 4;                        // lane half (wave32)

    v8f acc0, acc1, acc2, acc3;
    if (Cin) {
        const float* c0 = Cin + (size_t)m0 * N + n0 + l16;
#pragma unroll
        for (int j = 0; j < 8; ++j) {
            size_t r = (size_t)(j + kh * 8) * N;
            acc0[j] = c0[r];
            acc1[j] = c0[r + 16];
            acc2[j] = c0[r + 32];
            acc3[j] = c0[r + 48];
        }
    } else {
#pragma unroll
        for (int j = 0; j < 8; ++j) { acc0[j] = 0.f; acc1[j] = 0.f; acc2[j] = 0.f; acc3[j] = 0.f; }
    }

    const __bf16* ap  = A  + (size_t)(m0 + l16) * Kp;            // A row m
    const __bf16* bp0 = Wt + (size_t)(n0 + l16) * Kp;            // W rows n0+l16 + s*16
    const size_t  bstep = (size_t)16 * Kp;

    for (int k = 0; k < Kp; k += 32) {
        Frag a, b0, b1, b2, b3;
        // A 16x32 bf16 layout: lane-half kh holds K[k+kh*16 .. k+kh*16+15]
        a.u[0] = *(const uint4*)(ap + k + kh * 16);
        a.u[1] = *(const uint4*)(ap + k + kh * 16 + 8);
        // B 32x16 bf16 layout: lane-half kh holds K[k+kh*16 .. k+kh*16+15]
        const __bf16* bk = bp0 + k + kh * 16;
        b0.u[0] = *(const uint4*)(bk);
        b0.u[1] = *(const uint4*)(bk + 8);
        b1.u[0] = *(const uint4*)(bk + bstep);
        b1.u[1] = *(const uint4*)(bk + bstep + 8);
        b2.u[0] = *(const uint4*)(bk + 2 * bstep);
        b2.u[1] = *(const uint4*)(bk + 2 * bstep + 8);
        b3.u[0] = *(const uint4*)(bk + 3 * bstep);
        b3.u[1] = *(const uint4*)(bk + 3 * bstep + 8);
        acc0 = __builtin_amdgcn_wmma_f32_16x16x32_bf16(false, a.v, false, b0.v, (short)0, acc0, false, false);
        acc1 = __builtin_amdgcn_wmma_f32_16x16x32_bf16(false, a.v, false, b1.v, (short)0, acc1, false, false);
        acc2 = __builtin_amdgcn_wmma_f32_16x16x32_bf16(false, a.v, false, b2.v, (short)0, acc2, false, false);
        acc3 = __builtin_amdgcn_wmma_f32_16x16x32_bf16(false, a.v, false, b3.v, (short)0, acc3, false, false);
    }

#pragma unroll
    for (int s = 0; s < 4; ++s) {
        v8f acc = (s == 0) ? acc0 : (s == 1) ? acc1 : (s == 2) ? acc2 : acc3;
        const int n = n0 + s * 16 + l16;
        const float bv = bias ? bias[n] : 0.0f;
#pragma unroll
        for (int j = 0; j < 8; ++j) {
            int m = m0 + j + kh * 8;
            float v = acc[j] + bv;
            if (tok) {
                int b_ = m & 63, t_ = m >> 6;
                v *= (tok[b_ * SS + t_] > 0) ? 1.0f : 0.0f;
            }
            Cout[(size_t)m * N + n] = v;
        }
    }
}

// ---------------------------------------------------------------------------
// LSTM cell elementwise; gates row pitch GGP, PyTorch order i,f,g,o
// ---------------------------------------------------------------------------
__device__ __forceinline__ float sigf(float x) { return 1.0f / (1.0f + expf(-x)); }

__global__ void lstm_cell_kernel(const float* __restrict__ g, const float* __restrict__ bias,
                                 float* __restrict__ c, __bf16* __restrict__ hcur,
                                 __bf16* __restrict__ hhist, int t, int dirOff) {
    int idx = blockIdx.x * blockDim.x + threadIdx.x;
    if (idx >= BZ * HH) return;
    int b = idx / HH, j = idx % HH;
    const float* gb = g + (size_t)b * GGP;
    float gi = gb[j]          + bias[j];
    float gf = gb[HH + j]     + bias[HH + j];
    float gg = gb[2 * HH + j] + bias[2 * HH + j];
    float go = gb[3 * HH + j] + bias[3 * HH + j];
    float cn = sigf(gf) * c[idx] + sigf(gi) * tanhf(gg);
    float h  = sigf(go) * tanhf(cn);
    c[idx] = cn;
    hcur[(size_t)b * HP + j] = (__bf16)h;                          // next-step WMMA A operand
    hhist[(size_t)(t * BZ + b) * D2P + dirOff + j] = (__bf16)h;    // layer output (bf16 history)
}

// ---------------------------------------------------------------------------
// CRF: per-batch DP over S steps, NT=64 tags. One 64-thread block per batch.
// ---------------------------------------------------------------------------
__global__ void __launch_bounds__(64)
crf_kernel(const float* __restrict__ emit, const float* __restrict__ trans,
           const int* __restrict__ x, const int* __restrict__ y0,
           float* __restrict__ out) {
    const int b = blockIdx.x;
    const int i = threadIdx.x;                  // tag index 0..63
    __shared__ float T[NT * NT];
    __shared__ float score[NT];
    __shared__ float red[NT];

    for (int j = 0; j < NT; ++j) T[i * NT + j] = trans[i * NT + j];
    score[i] = (i == 2) ? 0.0f : -10000.0f;     // SOS = 2
    __syncthreads();

    for (int t = 0; t < SS; ++t) {
        float e = emit[((size_t)t * BZ + b) * NT + i];   // already masked
        int   m = x[b * SS + t] > 0;
        float mx = -3.4e38f;
        for (int j = 0; j < NT; ++j)
            mx = fmaxf(mx, score[j] + T[i * NT + j]);
        float s = 0.0f;
        for (int j = 0; j < NT; ++j)
            s += expf(score[j] + T[i * NT + j] - mx);
        float nv = e + mx + logf(s);
        __syncthreads();
        if (m) score[i] = nv;
        __syncthreads();
    }

    // gold path: each thread sums a strided subset of timesteps
    float gp = 0.0f;
    for (int t = i; t < SS; t += NT) {
        int yt = y0[b * SS + t];
        int yp = (t == 0) ? 2 : y0[b * SS + t - 1];
        float m = (x[b * SS + t] > 0) ? 1.0f : 0.0f;
        gp += emit[((size_t)t * BZ + b) * NT + yt] + T[yt * NT + yp] * m;
    }
    red[i] = gp;
    __syncthreads();

    if (i == 0) {
        float mx = -3.4e38f;
        for (int j = 0; j < NT; ++j) mx = fmaxf(mx, score[j]);
        float s = 0.0f;
        for (int j = 0; j < NT; ++j) s += expf(score[j] - mx);
        float Z = mx + logf(s);
        float gold = 0.0f;
        for (int j = 0; j < NT; ++j) gold += red[j];
        out[b] = Z - gold;
    }
}

// ---------------------------------------------------------------------------
// host orchestration
// ---------------------------------------------------------------------------
extern "C" void kernel_launch(void* const* d_in, const int* in_sizes, int n_in,
                              void* d_out, int out_size, void* d_ws, size_t ws_size,
                              hipStream_t stream) {
    (void)in_sizes; (void)n_in; (void)out_size; (void)ws_size;
    const int*   x      = (const int*)d_in[0];
    const int*   y0     = (const int*)d_in[1];
    const float* embed  = (const float*)d_in[2];
    const float* wih_f32[2][2] = { { (const float*)d_in[3],  (const float*)d_in[6]  },
                                   { (const float*)d_in[9],  (const float*)d_in[12] } };
    const float* whh_f32[2][2] = { { (const float*)d_in[4],  (const float*)d_in[7]  },
                                   { (const float*)d_in[10], (const float*)d_in[13] } };
    const float* bias_g[2][2]  = { { (const float*)d_in[5],  (const float*)d_in[8]  },
                                   { (const float*)d_in[11], (const float*)d_in[14] } };
    const float* W_out  = (const float*)d_in[15];
    const float* b_out  = (const float*)d_in[16];
    const float* trans  = (const float*)d_in[17];

    // ---- workspace layout (≈240 MB) ----
    char* base = (char*)d_ws; size_t off = 0;
    auto alloc = [&](size_t bytes) -> void* {
        off = (off + 255) & ~(size_t)255;
        void* p = base + off; off += bytes; return p;
    };
    __bf16* wihb[2][2]; __bf16* whhb[2][2];
    const int Kin[2] = { EP, D2P };   // padded input dims per layer
    for (int l = 0; l < 2; ++l)
        for (int d = 0; d < 2; ++d) {
            wihb[l][d] = (__bf16*)alloc((size_t)GGP * Kin[l] * 2);
            whhb[l][d] = (__bf16*)alloc((size_t)GGP * HP * 2);
        }
    __bf16* woutb = (__bf16*)alloc((size_t)NT * D2P * 2);
    __bf16* x0b   = (__bf16*)alloc((size_t)MM * EP * 2);
    __bf16* h1b   = (__bf16*)alloc((size_t)MM * D2P * 2);
    __bf16* h2b   = (__bf16*)alloc((size_t)MM * D2P * 2);
    float*  gpre  = (float*)alloc((size_t)MM * GGP * 4);
    float*  gstep = (float*)alloc((size_t)BZ * GGP * 4);
    __bf16* hcur  = (__bf16*)alloc((size_t)BZ * HP * 2);
    float*  cbuf  = (float*)alloc((size_t)BZ * HH * 4);
    float*  emitb = (float*)alloc((size_t)MM * NT * 4);

    const int KinRaw[2] = { EE, D2 };

    // ---- weight conversion to padded bf16 (N padded to GGP, K padded to 32x) ----
    for (int l = 0; l < 2; ++l)
        for (int d = 0; d < 2; ++d) {
            cvt_pad_bf16_kernel<<<1024, 256, 0, stream>>>(wih_f32[l][d], wihb[l][d], GG, GGP, KinRaw[l], Kin[l]);
            cvt_pad_bf16_kernel<<<1024, 256, 0, stream>>>(whh_f32[l][d], whhb[l][d], GG, GGP, HH, HP);
        }
    cvt_pad_bf16_kernel<<<256, 256, 0, stream>>>(W_out, woutb, NT, NT, D2, D2P);

    // ---- embedding gather + clear bf16 histories (zero pads persist) ----
    embed_gather_kernel<<<2048, 256, 0, stream>>>(x, embed, x0b);
    zero_u32_kernel<<<2048, 256, 0, stream>>>((uint32_t*)h1b, (size_t)MM * D2P / 2);
    zero_u32_kernel<<<2048, 256, 0, stream>>>((uint32_t*)h2b, (size_t)MM * D2P / 2);

    // ---- one bidirectional-LSTM direction pass ----
    auto runDir = [&](const __bf16* xin, int KinP, const __bf16* wih, const __bf16* whh,
                      const float* bias, __bf16* hout, int dirOff, bool rev) {
        // bulk input-gate GEMM: gpre[MM,GGP] = xin @ wih^T
        int waves = (MM / 16) * (GGP / 64);              // 32768
        gemm_bf16_wmma_kernel<<<waves / 4, 128, 0, stream>>>(
            xin, wih, nullptr, gpre, nullptr, nullptr, MM, GGP, KinP);
        // reset recurrent state
        zero_u32_kernel<<<64, 256, 0, stream>>>((uint32_t*)hcur, (size_t)BZ * HP / 2);
        zero_u32_kernel<<<128, 256, 0, stream>>>((uint32_t*)cbuf, (size_t)BZ * HH);
        // sequential recurrence
        for (int step = 0; step < SS; ++step) {
            int t = rev ? (SS - 1 - step) : step;
            // gstep = gpre[t] + hcur @ whh^T      (C-preload fuses the add)
            gemm_bf16_wmma_kernel<<<(BZ / 16) * (GGP / 64) / 4, 128, 0, stream>>>(
                hcur, whh, gpre + (size_t)t * BZ * GGP, gstep,
                nullptr, nullptr, BZ, GGP, HP);
            lstm_cell_kernel<<<(BZ * HH + 255) / 256, 256, 0, stream>>>(
                gstep, bias, cbuf, hcur, hout, t, dirOff);
        }
    };

    runDir(x0b, EP,  wihb[0][0], whhb[0][0], bias_g[0][0], h1b, 0,  false);
    runDir(x0b, EP,  wihb[0][1], whhb[0][1], bias_g[0][1], h1b, HH, true);
    runDir(h1b, D2P, wihb[1][0], whhb[1][0], bias_g[1][0], h2b, 0,  false);
    runDir(h1b, D2P, wihb[1][1], whhb[1][1], bias_g[1][1], h2b, HH, true);

    // ---- emission: emit[MM,NT] = (h2 @ W_out^T + b_out) * mask ----
    {
        int waves = (MM / 16) * (NT / 64);               // 1024
        gemm_bf16_wmma_kernel<<<waves / 4, 128, 0, stream>>>(
            h2b, woutb, nullptr, emitb, b_out, x, MM, NT, D2P);
    }

    // ---- CRF DP + gold path -> NLL per batch ----
    crf_kernel<<<BZ, NT, 0, stream>>>(emitb, trans, x, y0, (float*)d_out);
}
